// MultiHeadAttention_20031727468709
// MI455X (gfx1250) — compile-verified, matrix operand order checked by reading, not run on hardware
//
#include <hip/hip_runtime.h>
#include <hip/hip_bf16.h>

// ---------------------------------------------------------------------------
// MHA forward for MI455X (gfx1250, wave32, WMMA 16x16x32 f16 -> f32 accum).
// Compute-bound (~69 GFLOP vs ~100MB HBM traffic) => everything on the WMMA
// path; attention fused flash-style so S*S scores never hit HBM.
// Tensor Data Mover (tensor_load_to_lds + s_wait_tensorcnt, 6-arg clang-23
// builtin form) stages the flash K-tile and the f16 GEMM A-tile; prefetch
// hints (global_prefetch_b8) cover the f32 staging paths.
// ---------------------------------------------------------------------------

#define BB  2
#define SS  2048
#define DD  1024
#define HH  16
#define DKK 64
#define MM  (BB * SS)   // 4096 rows in the projection GEMMs

typedef __attribute__((ext_vector_type(16))) _Float16 v16h;
typedef __attribute__((ext_vector_type(8)))  _Float16 v8h;
typedef __attribute__((ext_vector_type(8)))  float    v8f;

// Tensor Data Mover availability.
#if defined(__AMDGCN__) && __has_builtin(__builtin_amdgcn_tensor_load_to_lds) && \
    __has_builtin(__builtin_amdgcn_s_wait_tensorcnt)
#define USE_TDM 1
#else
#define USE_TDM 0
#endif

#if USE_TDM
typedef __attribute__((ext_vector_type(4))) unsigned int u32x4;
typedef __attribute__((ext_vector_type(8))) int          i32x8;
typedef __attribute__((ext_vector_type(4))) int          i32x4;

// Build a D# (descriptor) per CDNA5 ISA and
// issue TENSOR_LOAD_TO_LDS. data_size fixed at 2 bytes (f16). Dimensions in
// element units. 1-D transfers: tile_d1 = 0.
__device__ __forceinline__ void tdm_load_f16(unsigned int lds_byte, const void* gsrc,
                                             unsigned int tensor_d0, unsigned int tensor_d1,
                                             unsigned int tile_d0, unsigned int tile_d1,
                                             unsigned long long stride0) {
    unsigned long long ga = (unsigned long long)(uintptr_t)gsrc;
    u32x4 g0;
    g0.x = 1u;                                                    // count=1, user mode
    g0.y = lds_byte;                                              // lds_addr [63:32]
    g0.z = (unsigned int)ga;                                      // global_addr [95:64]
    g0.w = (unsigned int)((ga >> 32) & 0x1FFFFFFull) | (2u << 30);// addr[56:32] | type=2
    i32x8 g1;
    g1[0] = (int)(1u << 16);                                      // data_size=1 -> 2B
    g1[1] = (int)((tensor_d0 & 0xFFFFu) << 16);                   // tensor_dim0[15:0]
    g1[2] = (int)(((tensor_d0 >> 16) & 0xFFFFu) |
                  ((tensor_d1 & 0xFFFFu) << 16));                 // td0[31:16]|td1[15:0]
    g1[3] = (int)(((tensor_d1 >> 16) & 0xFFFFu) |
                  ((tile_d0 & 0xFFFFu) << 16));                   // td1[31:16]|tile0
    g1[4] = (int)(tile_d1 & 0xFFFFu);                             // tile1 | tile2=0
    g1[5] = (int)(unsigned int)(stride0 & 0xFFFFFFFFull);         // dim0_stride[31:0]
    g1[6] = (int)(unsigned int)((stride0 >> 32) & 0xFFFFull);     // stride[47:32]|s1=0
    g1[7] = 0;
    i32x4 gz4 = {0, 0, 0, 0};
    i32x8 gz8 = {0, 0, 0, 0, 0, 0, 0, 0};
    __builtin_amdgcn_tensor_load_to_lds(g0, g1, gz4, gz4, gz8, 0);
}
#endif

// Generic shared pointer -> LDS byte address: flat LDS aperture maps by
// truncation to the low 32 bits (ISA sec 10.2).
__device__ __forceinline__ unsigned int lds_byte_addr(const void* p) {
    return (unsigned int)(uintptr_t)p;
}

// 16x32 f16 A-fragment / 32x16 f16 B-fragment: per ISA layout, lane L holds
// row/col (L&15); halves h map to k = (h>>3)*16 + (L>>4)*8 + (h&7), i.e. two
// CONTIGUOUS 8-half (16B) chunks at k-offsets (L>>4)*8 and 16+(L>>4)*8.
union Frag16 {
    v16h v;
    v8h  h[2];
};

__device__ __forceinline__ v8f wmma_f16(const Frag16& a, const Frag16& b, v8f c) {
    return __builtin_amdgcn_wmma_f32_16x16x32_f16(
        /*neg_a=*/false, a.v, /*neg_b=*/false, b.v,
        /*c_mod=*/(short)0, c, /*reuse_a=*/false, /*reuse_b=*/false);
}

// ---------------------------------------------------------------------------
// Tiled GEMM: C[M=4096, N=1024] = A[4096,1024] @ W[1024,1024]
// Block tile 128x128, 8 waves in a 4(M) x 2(N) grid, each wave 32x64
// (2x4 = 8 accumulators), K-step 32.
// out_mode 0: f32 row-major + bias -> d_out.  out_mode 1: f16 [B,H,S,DK].
// f16 A input is staged with the Tensor Data Mover when available.
// ---------------------------------------------------------------------------
__global__ __launch_bounds__(256)
void gemm128_wmma(const void* __restrict__ Aptr, int a_is_f16,
                  const float* __restrict__ W, const float* __restrict__ bias,
                  void* __restrict__ Cptr, int out_mode, float scale) {
    __shared__ _Float16 ldsA[128 * 32];   // A tile, row-major [m][k]
    __shared__ _Float16 ldsBt[128 * 32];  // W tile, transposed [n][k]

    const int t     = threadIdx.x;
    const int lane  = t & 31;
    const int wave  = t >> 5;
    const int l16   = lane & 15;
    const int hi    = lane >> 4;
    const int wm    = wave >> 1;      // 0..3 -> M offset wm*32
    const int wn    = wave & 1;       // 0..1 -> N offset wn*64
    const int mbase = blockIdx.x * 128;
    const int nbase = blockIdx.y * 128;

    v8f acc[8] = {};                  // [mi][ni] -> acc[mi*4+ni]

    for (int k0 = 0; k0 < DD; k0 += 32) {
        __syncthreads();
        // ---- stage A tile (128x32) as f16, row-major ----
        if (a_is_f16) {
#if USE_TDM
            if (t < 32) {   // one wave issues the DMA; TDM ignores EXEC
                tdm_load_f16(lds_byte_addr(ldsA),
                             (const _Float16*)Aptr + (size_t)mbase * DD + k0,
                             /*tensor_d0=*/DD, /*tensor_d1=*/(unsigned)(MM - mbase),
                             /*tile_d0=*/32, /*tile_d1=*/128, /*stride0=*/DD);
                __builtin_amdgcn_s_wait_tensorcnt((short)0);
            }
#else
            const int arow = t >> 1;
            const int acol = (t & 1) * 16;
            const _Float16* Af = (const _Float16*)Aptr;
            const _Float16* src = Af + (size_t)(mbase + arow) * DD + k0 + acol;
            _Float16* dst = ldsA + arow * 32 + acol;
            *(v8h*)(dst)     = *(const v8h*)(src);
            *(v8h*)(dst + 8) = *(const v8h*)(src + 8);
#endif
        } else {
            const int arow = t >> 1;
            const int acol = (t & 1) * 16;
            const float* Af = (const float*)Aptr;
            const float* src = Af + (size_t)(mbase + arow) * DD + k0 + acol;
            _Float16* dst = ldsA + arow * 32 + acol;
            if (k0 + 32 < DD) __builtin_prefetch(src + 32, 0, 1);
#pragma unroll
            for (int i = 0; i < 16; ++i) dst[i] = (_Float16)src[i];
        }
        // ---- stage W tile (32x128) transposed -> ldsBt[n][k] ----
        {
            const int bk = t >> 3;           // 0..31
            const int bn = (t & 7) * 16;     // 0..112
            const float* src = W + (size_t)(k0 + bk) * DD + nbase + bn;
            if (k0 + 32 < DD) __builtin_prefetch(src + (size_t)32 * DD, 0, 1);
#pragma unroll
            for (int i = 0; i < 16; ++i)
                ldsBt[(bn + i) * 32 + bk] = (_Float16)src[i];
        }
        __syncthreads();

        // ---- fragments + 8 WMMA ----
        Frag16 af[2], bf[4];
#pragma unroll
        for (int mi = 0; mi < 2; ++mi) {
            const _Float16* ar = ldsA + (wm * 32 + mi * 16 + l16) * 32;
            af[mi].h[0] = *(const v8h*)(ar + hi * 8);
            af[mi].h[1] = *(const v8h*)(ar + 16 + hi * 8);
        }
#pragma unroll
        for (int ni = 0; ni < 4; ++ni) {
            const _Float16* br = ldsBt + (wn * 64 + ni * 16 + l16) * 32;
            bf[ni].h[0] = *(const v8h*)(br + hi * 8);
            bf[ni].h[1] = *(const v8h*)(br + 16 + hi * 8);
        }
#pragma unroll
        for (int mi = 0; mi < 2; ++mi)
#pragma unroll
            for (int ni = 0; ni < 4; ++ni)
                acc[mi * 4 + ni] = wmma_f16(af[mi], bf[ni], acc[mi * 4 + ni]);
    }

    // ---- epilogue: C layout row M = r + 8*(lane>>4), col N = lane&15 ----
#pragma unroll
    for (int mi = 0; mi < 2; ++mi)
#pragma unroll
        for (int ni = 0; ni < 4; ++ni)
#pragma unroll
            for (int r = 0; r < 8; ++r) {
                const int m = mbase + wm * 32 + mi * 16 + r + 8 * hi;
                const int n = nbase + wn * 64 + ni * 16 + l16;
                const float v = acc[mi * 4 + ni][r] * scale;
                if (out_mode == 0) {
                    ((float*)Cptr)[(size_t)m * DD + n] = v + bias[n];
                } else {
                    const int b = m >> 11, s = m & (SS - 1);
                    const int h = n >> 6,  dk = n & (DKK - 1);
                    ((_Float16*)Cptr)[(((size_t)(b * HH + h)) * SS + s) * DKK + dk] =
                        (_Float16)v;
                }
            }
}

// ---------------------------------------------------------------------------
// Fused causal flash attention. Q already carries 1/sqrt(DK).
// Grid: x = q-tile (S/64 = 32), y = b*H + h (32). Block = 128 (4 waves);
// wave w owns q rows [qb*64 + w*16, +16).
// K tile is DMA'd into LDS by the TDM (one copy shared by all 4 waves).
// ---------------------------------------------------------------------------
__global__ __launch_bounds__(128)
void flash_attn(const _Float16* __restrict__ Qh, const _Float16* __restrict__ Kh,
                const _Float16* __restrict__ Vh, _Float16* __restrict__ Oh) {
    __shared__ _Float16 Vt[64 * 64];        // V block transposed [dk][key]
    __shared__ _Float16 Pl[4 * 16 * 64];    // per-wave P staging [q][key]
#if USE_TDM
    __shared__ _Float16 ldsK[64 * 64];      // K block row-major [key][dk]
#endif

    const int t    = threadIdx.x;
    const int lane = t & 31;
    const int wave = t >> 5;
    const int l16  = lane & 15;
    const int hi   = lane >> 4;
    const int qb   = blockIdx.x;
    const int bh   = blockIdx.y;
    const int b    = bh >> 4, h = bh & 15;
    const int qrow0 = qb * 64 + wave * 16;

    // Q A-fragments (16 q-rows x 64 dk = 2 frags), straight from global:
    // layout makes each fragment two contiguous 16B loads per lane.
    Frag16 qa[2];
    {
        const _Float16* qp = Qh + ((size_t)bh * SS + qrow0 + l16) * DKK;
#pragma unroll
        for (int f = 0; f < 2; ++f) {
            qa[f].h[0] = *(const v8h*)(qp + f * 32 + hi * 8);
            qa[f].h[1] = *(const v8h*)(qp + f * 32 + 16 + hi * 8);
        }
    }

    v8f   o[4] = {};                        // 16 q x 64 dk accumulators
    float mrow[8], lrow[8];
#pragma unroll
    for (int r = 0; r < 8; ++r) { mrow[r] = -1e30f; lrow[r] = 0.0f; }

    _Float16* Pw = Pl + wave * 16 * 64;

    for (int kb = 0; kb <= qb; ++kb) {
        __syncthreads();
        // ---- stage V block transposed into LDS ----
        {
            const int keyl = t >> 1;                    // 0..63
            const int dkb  = (t & 1) * 32;
            const _Float16* vsrc =
                Vh + ((size_t)bh * SS + kb * 64 + keyl) * DKK + dkb;
            if (kb < qb) __builtin_prefetch(vsrc + (size_t)64 * DKK, 0, 1);
            const v8h* vp = (const v8h*)vsrc;
#pragma unroll
            for (int j = 0; j < 4; ++j) {
                v8h x = vp[j];
#pragma unroll
                for (int i = 0; i < 8; ++i)
                    Vt[(dkb + j * 8 + i) * 64 + keyl] = x[i];
            }
        }
#if USE_TDM
        // ---- DMA the 8KB K block into LDS (flat 1-D descriptor) ----
        if (t < 32) {
            tdm_load_f16(lds_byte_addr(ldsK),
                         Kh + ((size_t)bh * SS + kb * 64) * DKK,
                         /*tensor_d0=*/64 * DKK, /*tensor_d1=*/1,
                         /*tile_d0=*/64 * DKK, /*tile_d1=*/0,
                         /*stride0=*/64 * DKK);
            __builtin_amdgcn_s_wait_tensorcnt((short)0);
        }
#endif
        __syncthreads();

        // ---- scores: S = Q @ K^T ----
        v8f sc[4] = {};
#pragma unroll
        for (int ni = 0; ni < 4; ++ni) {
#if USE_TDM
            const _Float16* kp = ldsK + (size_t)(ni * 16 + l16) * DKK;
#else
            const _Float16* kp =
                Kh + ((size_t)bh * SS + kb * 64 + ni * 16 + l16) * DKK;
#endif
#pragma unroll
            for (int f = 0; f < 2; ++f) {
                Frag16 kf;
                kf.h[0] = *(const v8h*)(kp + f * 32 + hi * 8);
                kf.h[1] = *(const v8h*)(kp + f * 32 + 16 + hi * 8);
                sc[ni] = wmma_f16(qa[f], kf, sc[ni]);
            }
        }

        // ---- causal mask + online softmax (C layout: row r+8*hi, col l16) --
#pragma unroll
        for (int r = 0; r < 8; ++r) {
            const int q = qrow0 + r + 8 * hi;
            float mx = -1e30f;
#pragma unroll
            for (int ni = 0; ni < 4; ++ni) {
                const int kcol = kb * 64 + ni * 16 + l16;
                float s = sc[ni][r];
                if (kcol > q) s = -1e30f;
                sc[ni][r] = s;
                mx = fmaxf(mx, s);
            }
            mx = fmaxf(mx, __shfl_xor(mx, 1, 32));
            mx = fmaxf(mx, __shfl_xor(mx, 2, 32));
            mx = fmaxf(mx, __shfl_xor(mx, 4, 32));
            mx = fmaxf(mx, __shfl_xor(mx, 8, 32));
            const float mnew  = fmaxf(mrow[r], mx);
            const float alpha = __expf(mrow[r] - mnew);
            mrow[r] = mnew;
            float sum = 0.0f;
#pragma unroll
            for (int ni = 0; ni < 4; ++ni) {
                const float p = __expf(sc[ni][r] - mnew);
                sc[ni][r] = p;
                sum += p;
            }
            sum += __shfl_xor(sum, 1, 32);
            sum += __shfl_xor(sum, 2, 32);
            sum += __shfl_xor(sum, 4, 32);
            sum += __shfl_xor(sum, 8, 32);
            lrow[r] = lrow[r] * alpha + sum;
#pragma unroll
            for (int ni = 0; ni < 4; ++ni) o[ni][r] *= alpha;
        }

        // ---- reshape P (C layout -> A layout) through per-wave LDS ----
#pragma unroll
        for (int ni = 0; ni < 4; ++ni)
#pragma unroll
            for (int r = 0; r < 8; ++r)
                Pw[(r + 8 * hi) * 64 + ni * 16 + l16] = (_Float16)sc[ni][r];
        // same-wave LDS ops are in-order; no barrier needed for private region
        Frag16 pa[2];
        {
            const _Float16* pp = Pw + l16 * 64;
#pragma unroll
            for (int f = 0; f < 2; ++f) {
                pa[f].h[0] = *(const v8h*)(pp + f * 32 + hi * 8);
                pa[f].h[1] = *(const v8h*)(pp + f * 32 + 16 + hi * 8);
            }
        }

        // ---- O += P @ V  (V B-frags from transposed LDS tile) ----
#pragma unroll
        for (int nd = 0; nd < 4; ++nd) {
            const _Float16* vr = Vt + (nd * 16 + l16) * 64;
#pragma unroll
            for (int f = 0; f < 2; ++f) {
                Frag16 vf;
                vf.h[0] = *(const v8h*)(vr + f * 32 + hi * 8);
                vf.h[1] = *(const v8h*)(vr + f * 32 + 16 + hi * 8);
                o[nd] = wmma_f16(pa[f], vf, o[nd]);
            }
        }
    }

    // ---- normalize + merge heads: Oh[B,S,D] f16 ----
#pragma unroll
    for (int r = 0; r < 8; ++r) {
        const int q = qrow0 + r + 8 * hi;
        const float inv = 1.0f / lrow[r];
#pragma unroll
        for (int nd = 0; nd < 4; ++nd) {
            const int dk = nd * 16 + l16;
            Oh[((size_t)(b * SS + q)) * DD + h * DKK + dk] =
                (_Float16)(o[nd][r] * inv);
        }
    }
}

// ---------------------------------------------------------------------------
extern "C" void kernel_launch(void* const* d_in, const int* in_sizes, int n_in,
                              void* d_out, int out_size, void* d_ws, size_t ws_size,
                              hipStream_t stream) {
    const float* query = (const float*)d_in[0];
    const float* key   = (const float*)d_in[1];
    const float* value = (const float*)d_in[2];
    const float* Wq    = (const float*)d_in[3];
    const float* Wk    = (const float*)d_in[4];
    const float* Wv    = (const float*)d_in[5];
    const float* Wp    = (const float*)d_in[6];
    const float* bp    = (const float*)d_in[7];
    float* out = (float*)d_out;

    const size_t headElems = (size_t)BB * HH * SS * DKK;  // 4M halves each
    _Float16* Qh = (_Float16*)d_ws;
    _Float16* Kh = Qh + headElems;
    _Float16* Vh = Kh + headElems;
    _Float16* Oh = Vh + headElems;                        // [B,S,D] f16

    dim3 gGrid(MM / 128, DD / 128);   // 32 x 8
    const float qscale = 0.125f;      // 1/sqrt(DK) folded into Q

    gemm128_wmma<<<gGrid, 256, 0, stream>>>(query, 0, Wq, nullptr, Qh, 1, qscale);
    gemm128_wmma<<<gGrid, 256, 0, stream>>>(key,   0, Wk, nullptr, Kh, 1, 1.0f);
    gemm128_wmma<<<gGrid, 256, 0, stream>>>(value, 0, Wv, nullptr, Vh, 1, 1.0f);

    dim3 aGrid(SS / 64, BB * HH);     // 32 x 32
    flash_attn<<<aGrid, 128, 0, stream>>>(Qh, Kh, Vh, Oh);

    gemm128_wmma<<<gGrid, 256, 0, stream>>>(Oh, 1, Wp, bp, out, 0, 1.0f);
}
